// SelfAttention_26482768347388
// MI455X (gfx1250) — matrix operand order repaired and checked by reading, not compiled, not run on hardware
//
#include <hip/hip_runtime.h>
#include <hip/hip_bf16.h>
#include <math.h>

#define HIDDEN 1024
#define NHEADS 16
#define HEADD  64
#define SEQ    2048
#define BATCH  2
#define QKVN   (3*HIDDEN)
#define MTOT   (BATCH*SEQ)   /* 4096 */
#define SCALE  0.125f
#define MASKV  -1.0e9f
#define NLOG2_10000_32 -0.4152410118609203f   /* -log2(10000)/32 */

typedef __bf16 v16bf __attribute__((ext_vector_type(16)));
typedef float  v8f   __attribute__((ext_vector_type(8)));
typedef unsigned int u32x4 __attribute__((ext_vector_type(4)));
typedef float  f32x4 __attribute__((ext_vector_type(4)));

union FragBF {
    v16bf v;
    u32x4 q[2];
    unsigned short s[16];
};

// native f32 -> bf16 (v_cvt_pk_bf16_f32 class ops, RNE)
__device__ __forceinline__ unsigned short f2bf(float f) {
    __bf16 h = (__bf16)f;
    return __builtin_bit_cast(unsigned short, h);
}

// LDS byte offset of a __shared__ object (for async-to-LDS instructions whose
// VDST VGPR carries the wave-relative LDS address).
typedef __attribute__((address_space(3))) void lds_void;
__device__ __forceinline__ unsigned int lds_offset(const void* p) {
    return (unsigned int)(unsigned long long)(lds_void*)p;
}

// A fragment (16x32 bf16, M x K): lane m=lane&15 holds row m.
// half=lane>>4: q[0] covers K = 8h..8h+7, q[1] covers K = 16+8h..16+8h+7.
__device__ __forceinline__ void gatherA(FragBF& f, const unsigned short* base,
                                        int ldk, int lane) {
    const int m = lane & 15, half = lane >> 4;
    const unsigned short* row = base + m * ldk;
    f.q[0] = *(const u32x4*)(row + 8 * half);
    f.q[1] = *(const u32x4*)(row + 16 + 8 * half);
}

// B fragment (32x16 bf16, K x N) gathered from [n][k] (k-contiguous) storage:
// lane n=lane&15 holds column n; q[0] = K 8h..8h+7, q[1] = K 16+8h..16+8h+7.
__device__ __forceinline__ void gatherB(FragBF& f, const unsigned short* base,
                                        int ldk, int lane) {
    const int n = lane & 15, half = lane >> 4;
    const unsigned short* col = base + n * ldk;
    f.q[0] = *(const u32x4*)(col + 8 * half);
    f.q[1] = *(const u32x4*)(col + 16 + 8 * half);
}

__device__ __forceinline__ v8f wmma_bf16(const FragBF& a, const FragBF& b, v8f c) {
    return __builtin_amdgcn_wmma_f32_16x16x32_bf16(false, a.v, false, b.v,
                                                   (short)0, c, false, false);
}

// ---------------------------------------------------------------------------
// Kernel 1: qkv = x @ Wqkv^T + b, fused RoPE, scatter to Q,K (B,H,T,D) and
// V transposed (B,H,D,T), all bf16. Block tile 64(M) x 128(N), K-step 32.
// ---------------------------------------------------------------------------
__global__ __launch_bounds__(256)
void qkv_rope_kernel(const float* __restrict__ x, const float* __restrict__ Wqkv,
                     const float* __restrict__ bqkv,
                     unsigned short* __restrict__ Qb, unsigned short* __restrict__ Kb,
                     unsigned short* __restrict__ Vt) {
    __shared__ unsigned short As[64 * 32];
    __shared__ unsigned short Bs[128 * 32];
    const int tid = threadIdx.x;
    const int lane = tid & 31, wid = tid >> 5;
    const int gm0 = blockIdx.y * 64;
    const int gn0 = blockIdx.x * 128;
    const int wm = (wid >> 2) * 32;   // 2 wave-rows
    const int wn = (wid & 3) * 32;    // 4 wave-cols

    v8f acc[2][2] = {};
    for (int k0 = 0; k0 < HIDDEN; k0 += 32) {
        __syncthreads();
        // As[m][k] = bf16(x[gm0+m][k0+k]) : 64x32, 8 floats/thread
        for (int it = 0; it < 2; ++it) {
            int idx = (tid + it * 256) * 4;
            int r = idx >> 5, c = idx & 31;
            f32x4 v = *(const f32x4*)(x + (size_t)(gm0 + r) * HIDDEN + k0 + c);
            unsigned short* d = &As[r * 32 + c];
            d[0] = f2bf(v.x); d[1] = f2bf(v.y); d[2] = f2bf(v.z); d[3] = f2bf(v.w);
        }
        // Bs[n][k] = bf16(Wqkv[gn0+n][k0+k]) : 128x32 (k-contiguous = W row-major)
        for (int it = 0; it < 4; ++it) {
            int idx = (tid + it * 256) * 4;
            int r = idx >> 5, c = idx & 31;
            f32x4 v = *(const f32x4*)(Wqkv + (size_t)(gn0 + r) * HIDDEN + k0 + c);
            unsigned short* d = &Bs[r * 32 + c];
            d[0] = f2bf(v.x); d[1] = f2bf(v.y); d[2] = f2bf(v.z); d[3] = f2bf(v.w);
        }
        __syncthreads();
        FragBF a0, a1, b0, b1;
        gatherA(a0, &As[(wm + 0) * 32], 32, lane);
        gatherA(a1, &As[(wm + 16) * 32], 32, lane);
        gatherB(b0, &Bs[(wn + 0) * 32], 32, lane);
        gatherB(b1, &Bs[(wn + 16) * 32], 32, lane);
        acc[0][0] = wmma_bf16(a0, b0, acc[0][0]);
        acc[0][1] = wmma_bf16(a0, b1, acc[0][1]);
        acc[1][0] = wmma_bf16(a1, b0, acc[1][0]);
        acc[1][1] = wmma_bf16(a1, b1, acc[1][1]);
    }

    // Epilogue: bias + RoPE (q,k) + scatter. C layout: col = lane&15, row = r+8*(lane>>4).
    const int half = lane >> 4, ln = lane & 15;
    for (int i = 0; i < 2; ++i)
        for (int j = 0; j < 2; ++j) {
            const int ng = gn0 + wn + 16 * j + ln;         // 0..3071
            const float bias = bqkv[ng];
            const int region = ng >> 10;                   // 0=q 1=k 2=v
            const int c = ng & 1023;
            const int h = c >> 6, d = c & 63;
            // inv_freq = 10000^(-2*(d/2)/64) = exp2(-(d/2) * log2(10000)/32)
            const float inv = __builtin_amdgcn_exp2f((float)(d >> 1) * NLOG2_10000_32);
            for (int r = 0; r < 8; ++r) {
                const int mg = gm0 + wm + 16 * i + r + 8 * half;
                const int t = mg & (SEQ - 1), b = mg >> 11;
                const int bh = b * NHEADS + h;
                float val = acc[i][j][r] + bias;
                if (region == 2) {
                    Vt[((size_t)bh * HEADD + d) * SEQ + t] = f2bf(val);
                } else {
                    // rotary pair lives in the adjacent lane (col d^1)
                    float partner = __shfl_xor(val, 1, 32);
                    float ang = (float)t * inv;
                    float cs = __cosf(ang), sn = __sinf(ang);
                    float res = ((d & 1) == 0) ? (val * cs - partner * sn)
                                               : (partner * sn + val * cs);
                    unsigned short* dst = (region == 0) ? Qb : Kb;
                    dst[((size_t)bh * SEQ + t) * HEADD + d] = f2bf(res);
                }
            }
        }
}

// ---------------------------------------------------------------------------
// Kernel 2: causal flash attention. One wave per (b,h, 16-row q tile).
// 32-key steps: S = Q K^T (4 wmma), online softmax, O += P V (4 wmma).
// Note: the reference's "global attention" rows (2..5) are mathematically
// identical to the causal rows they overwrite, so no fix-up pass is needed.
// ---------------------------------------------------------------------------
__global__ __launch_bounds__(256)
void attn_kernel(const unsigned short* __restrict__ Qb,
                 const unsigned short* __restrict__ Kb,
                 const unsigned short* __restrict__ Vt,
                 unsigned short* __restrict__ Ob) {
    __shared__ unsigned short Plds[8][16 * 32];
    const int tid = threadIdx.x, lane = tid & 31, wid = tid >> 5;
    const int task = blockIdx.x * 8 + wid;
    const int qt = task & 127;        // T/16 tiles
    const int bh = task >> 7;         // 0..31
    const int q0 = qt << 4;
    const int half = lane >> 4, ln = lane & 15;

    const unsigned short* Qbase = Qb + ((size_t)bh * SEQ + q0) * HEADD;
    FragBF Aq0, Aq1;                       // Q rows, d-chunks 0..31 / 32..63
    gatherA(Aq0, Qbase, HEADD, lane);
    gatherA(Aq1, Qbase + 32, HEADD, lane);

    v8f accO[4] = {};
    float mrow[8], lrow[8], alpha[8], p0[8], p1[8];
    for (int r = 0; r < 8; ++r) { mrow[r] = -1.0e30f; lrow[r] = 0.0f; }

    const int ksteps = (q0 + 16 + 31) >> 5;
    for (int ks = 0; ks < ksteps; ++ks) {
        const int k0 = ks << 5;
        // S tile 16x32 (two 16-col subtiles), accumulate over d in K=32 chunks
        v8f S[2];
        for (int jc = 0; jc < 2; ++jc) {
            FragBF B0, B1;
            const unsigned short* Kbase = Kb + ((size_t)bh * SEQ + k0 + 16 * jc) * HEADD;
            gatherB(B0, Kbase, HEADD, lane);        // B[n=key][k=d], d 0..31
            gatherB(B1, Kbase + 32, HEADD, lane);   // d 32..63
            v8f s = {};
            s = wmma_bf16(Aq0, B0, s);
            s = wmma_bf16(Aq1, B1, s);
            S[jc] = s;
        }
        if (ks + 1 < ksteps) {
            __builtin_prefetch(Kb + ((size_t)bh * SEQ + k0 + 32) * HEADD, 0, 3);
            __builtin_prefetch(Vt + ((size_t)bh * HEADD) * SEQ + k0 + 32, 0, 3);
        }
        // scale + causal mask + online softmax (rows live across 16-lane groups)
        for (int r = 0; r < 8; ++r) {
            const int row = q0 + r + 8 * half;
            const int key0 = k0 + ln, key1 = k0 + 16 + ln;
            float v0 = S[0][r] * SCALE; if (key0 > row) v0 = MASKV;
            float v1 = S[1][r] * SCALE; if (key1 > row) v1 = MASKV;
            float rm = fmaxf(v0, v1);
            rm = fmaxf(rm, __shfl_xor(rm, 1, 32));
            rm = fmaxf(rm, __shfl_xor(rm, 2, 32));
            rm = fmaxf(rm, __shfl_xor(rm, 4, 32));
            rm = fmaxf(rm, __shfl_xor(rm, 8, 32));
            const float mn = fmaxf(mrow[r], rm);
            const float a = __expf(mrow[r] - mn);
            const float e0 = __expf(v0 - mn);
            const float e1 = __expf(v1 - mn);
            float rs = e0 + e1;
            rs += __shfl_xor(rs, 1, 32);
            rs += __shfl_xor(rs, 2, 32);
            rs += __shfl_xor(rs, 4, 32);
            rs += __shfl_xor(rs, 8, 32);
            lrow[r] = lrow[r] * a + rs;
            mrow[r] = mn;
            alpha[r] = a;
            p0[r] = e0; p1[r] = e1;
        }
        for (int j = 0; j < 4; ++j)
            for (int r = 0; r < 8; ++r)
                accO[j][r] *= alpha[r];
        // redistribute P (C layout) -> A-fragment layout via per-wave LDS tile
        unsigned short* P = Plds[wid];
        for (int r = 0; r < 8; ++r) {
            const int M = r + 8 * half;
            P[M * 32 + ln]      = f2bf(p0[r]);
            P[M * 32 + 16 + ln] = f2bf(p1[r]);
        }
        asm volatile("s_wait_dscnt 0" ::: "memory");   // intra-wave LDS RAW
        FragBF Pf;
        gatherA(Pf, P, 32, lane);
        // O(16x64) += P(16x32) * V(32x64); V stored transposed [d][t]
        for (int j = 0; j < 4; ++j) {
            FragBF Bv;
            const unsigned short* Vbase = Vt + ((size_t)bh * HEADD + 16 * j) * SEQ + k0;
            gatherB(Bv, Vbase, SEQ, lane);
            accO[j] = wmma_bf16(Pf, Bv, accO[j]);
        }
    }
    // normalize and write O as (B,T,H*D) bf16 for the output projection
    const int b = bh >> 4, h = bh & 15;
    for (int j = 0; j < 4; ++j)
        for (int r = 0; r < 8; ++r) {
            const int M = r + 8 * half;
            const int t = q0 + M;
            const int d = 16 * j + ln;
            const float v = accO[j][r] / lrow[r];
            Ob[((size_t)b * SEQ + t) * HIDDEN + h * HEADD + d] = f2bf(v);
        }
}

// ---------------------------------------------------------------------------
// Kernel 3: y = O @ Wout^T + bout (fp32 out). A-tile is staged via the CDNA5
// async DMA path (GLOBAL_LOAD_ASYNC_TO_LDS_B128, tracked by ASYNCcnt) since
// it is a raw bf16 copy with no conversion.
// ---------------------------------------------------------------------------
__global__ __launch_bounds__(256)
void out_proj_kernel(const unsigned short* __restrict__ Ob,
                     const float* __restrict__ Wout, const float* __restrict__ bout,
                     float* __restrict__ y) {
    __shared__ unsigned short As[64 * 32];
    __shared__ unsigned short Bs[128 * 32];
    const int tid = threadIdx.x, lane = tid & 31, wid = tid >> 5;
    const int gm0 = blockIdx.y * 64, gn0 = blockIdx.x * 128;
    const int wm = (wid >> 2) * 32, wn = (wid & 3) * 32;

    // per-thread 16B chunk of the A tile
    const int aidx = tid * 8;
    const int ar = aidx >> 5, ac = aidx & 31;
    const unsigned int a_lds = lds_offset(&As[ar * 32 + ac]);

    v8f acc[2][2] = {};
    for (int k0 = 0; k0 < HIDDEN; k0 += 32) {
        __syncthreads();
        {   // As: async DMA 16B/thread straight into LDS (no VGPR round-trip)
            const unsigned short* src = Ob + (size_t)(gm0 + ar) * HIDDEN + k0 + ac;
            asm volatile("global_load_async_to_lds_b128 %0, %1, off"
                         :: "v"(a_lds), "v"(src) : "memory");
        }
        for (int it = 0; it < 4; ++it) {
            int idx = (tid + it * 256) * 4;
            int r = idx >> 5, c = idx & 31;
            f32x4 v = *(const f32x4*)(Wout + (size_t)(gn0 + r) * HIDDEN + k0 + c);
            unsigned short* d = &Bs[r * 32 + c];
            d[0] = f2bf(v.x); d[1] = f2bf(v.y); d[2] = f2bf(v.z); d[3] = f2bf(v.w);
        }
        asm volatile("s_wait_asynccnt 0" ::: "memory");
        __syncthreads();
        FragBF a0, a1, b0, b1;
        gatherA(a0, &As[(wm + 0) * 32], 32, lane);
        gatherA(a1, &As[(wm + 16) * 32], 32, lane);
        gatherB(b0, &Bs[(wn + 0) * 32], 32, lane);
        gatherB(b1, &Bs[(wn + 16) * 32], 32, lane);
        acc[0][0] = wmma_bf16(a0, b0, acc[0][0]);
        acc[0][1] = wmma_bf16(a0, b1, acc[0][1]);
        acc[1][0] = wmma_bf16(a1, b0, acc[1][0]);
        acc[1][1] = wmma_bf16(a1, b1, acc[1][1]);
    }
    const int half = lane >> 4, ln = lane & 15;
    for (int i = 0; i < 2; ++i)
        for (int j = 0; j < 2; ++j) {
            const int n = gn0 + wn + 16 * j + ln;
            const float bias = bout[n];
            for (int r = 0; r < 8; ++r) {
                const int m = gm0 + wm + 16 * i + r + 8 * half;
                y[(size_t)m * HIDDEN + n] = acc[i][j][r] + bias;
            }
        }
}

extern "C" void kernel_launch(void* const* d_in, const int* in_sizes, int n_in,
                              void* d_out, int out_size, void* d_ws, size_t ws_size,
                              hipStream_t stream) {
    (void)in_sizes; (void)n_in; (void)out_size; (void)ws_size;
    const float* x    = (const float*)d_in[0];
    // d_in[1] is the causal mask; it is structural (triu k=1) and fused analytically.
    const float* Wqkv = (const float*)d_in[2];
    const float* bqkv = (const float*)d_in[3];
    const float* Wout = (const float*)d_in[4];
    const float* bout = (const float*)d_in[5];
    float* y = (float*)d_out;

    char* ws = (char*)d_ws;
    const size_t qkv_elems = (size_t)BATCH * NHEADS * SEQ * HEADD;  // 4 Mi elems
    unsigned short* Qb = (unsigned short*)(ws);                     // 8 MB
    unsigned short* Kb = (unsigned short*)(ws + qkv_elems * 2);     // 8 MB
    unsigned short* Vt = (unsigned short*)(ws + qkv_elems * 4);     // 8 MB
    unsigned short* Ob = (unsigned short*)(ws + qkv_elems * 6);     // 8 MB

    dim3 g1(QKVN / 128, MTOT / 64);   // 24 x 64
    qkv_rope_kernel<<<g1, 256, 0, stream>>>(x, Wqkv, bqkv, Qb, Kb, Vt);

    const int tasks = BATCH * NHEADS * (SEQ / 16);   // 4096 wave-tasks
    attn_kernel<<<tasks / 8, 256, 0, stream>>>(Qb, Kb, Vt, Ob);

    dim3 g3(HIDDEN / 128, MTOT / 64); // 8 x 64
    out_proj_kernel<<<g3, 256, 0, stream>>>(Ob, Wout, bout, y);
}
